// CaptionModel_30305289240916
// MI455X (gfx1250) — matrix operand order repaired and checked by reading, not compiled, not run on hardware
//
#include <hip/hip_runtime.h>
#include <math.h>

// ---------------- problem constants ----------------
constexpr int B   = 64;
constexpr int L   = 49;
constexpr int ENC = 2048;
constexpr int DEC = 512;
constexpr int EMB = 512;
constexpr int ATT = 256;
constexpr int V   = 10000;
constexpr int T   = 50;
constexpr int BL  = B * L;      // 3136
constexpr int G3  = 3 * DEC;    // 1536
constexpr int XIN = EMB + DEC;  // 1024

// ---------------- types for WMMA ----------------
typedef __bf16 bf16_t;
typedef bf16_t v16bf __attribute__((ext_vector_type(16)));
typedef float  v8f   __attribute__((ext_vector_type(8)));

struct U4 { unsigned int a, b, c, d; };           // 16 bytes = 8 bf16
union BFrag { unsigned short us[16]; U4 q[2]; v16bf v; };

__device__ __forceinline__ unsigned short f32_to_bf16(float f) {
    unsigned int u = __builtin_bit_cast(unsigned int, f);
    u += 0x7FFFu + ((u >> 16) & 1u);              // round-to-nearest-even
    return (unsigned short)(u >> 16);
}

// ---------------- generic WMMA GEMM, 32x16 macro-tile per wave ----------------
// C[M,N](f32) = A[M,K](bf16, row-major, lda) @ Bt[N,K](bf16, row-major, ldb)^T + bias[N]
// Each wave owns a 32(M) x 16(N) C macro-tile: two A fragments share one B fragment,
// so the K loop issues 2 x v_wmma_f32_16x16x32_bf16 per 3 x 32B fragment loads.
// Requires M%32==0, N%16==0, K%32==0 (true for all GEMMs here).
__global__ void wmma_gemm_bf16(const unsigned short* __restrict__ A, int lda,
                               const unsigned short* __restrict__ Bt, int ldb,
                               const float* __restrict__ bias,
                               float* __restrict__ C, long long ldc,
                               int tiles_m, int tiles_n, int K)
{
    int tile = blockIdx.x * 8 + (threadIdx.x >> 5);
    if (tile >= tiles_m * tiles_n) return;
    int tm   = tile % tiles_m;            // macro-row (32 rows)
    int tn   = tile / tiles_m;            // 16-col tile
    int lane = threadIdx.x & 31;
    int half = lane >> 4;                 // 0: lanes 0-15, 1: lanes 16-31
    int l16  = lane & 15;

    // A 16x32 bf16 frag: lane row; elems0-7 = K k+half*8.., elems8-15 = K k+16+half*8..
    const unsigned short* arow0 = A  + (size_t)(tm * 32 + l16) * lda + half * 8;
    const unsigned short* arow1 = arow0 + (size_t)16 * lda;
    // B 32x16 bf16 frag: column n = tn*16+l16 from Bt row n; elems0-7 = K k+half*8.., elems8-15 = K k+16+half*8..
    const unsigned short* brow  = Bt + (size_t)(tn * 16 + l16) * ldb + half * 8;

    v8f acc0 = {};
    v8f acc1 = {};
    for (int k = 0; k < K; k += 32) {
        BFrag a0, a1, b;
        a0.q[0] = *(const U4*)(arow0 + k);
        a0.q[1] = *(const U4*)(arow0 + k + 16);
        a1.q[0] = *(const U4*)(arow1 + k);
        a1.q[1] = *(const U4*)(arow1 + k + 16);
        b.q[0]  = *(const U4*)(brow + k);
        b.q[1]  = *(const U4*)(brow + k + 16);
        if (k + 64 < K) {
            __builtin_prefetch((const void*)(brow  + k + 64), 0, 1);
            __builtin_prefetch((const void*)(arow0 + k + 64), 0, 1);
            __builtin_prefetch((const void*)(arow1 + k + 64), 0, 1);
        }
        acc0 = __builtin_amdgcn_wmma_f32_16x16x32_bf16(
                   false, a0.v, false, b.v, (short)0, acc0, false, false);
        acc1 = __builtin_amdgcn_wmma_f32_16x16x32_bf16(
                   false, a1.v, false, b.v, (short)0, acc1, false, false);
    }

    int col = tn * 16 + l16;
    float bv = bias ? bias[col] : 0.0f;
    long long base0 = (long long)(tm * 32 + half * 8) * ldc + col;
    long long base1 = base0 + (long long)16 * ldc;
#pragma unroll
    for (int r = 0; r < 8; ++r) {
        C[base0 + (long long)r * ldc] = acc0[r] + bv;
        C[base1 + (long long)r * ldc] = acc1[r] + bv;
    }
}

// ---------------- conversion / init kernels ----------------
__global__ void cvt_copy_bf16(const float* __restrict__ src,
                              unsigned short* __restrict__ dst, int n) {
    int i = blockIdx.x * 256 + threadIdx.x;
    if (i < n) dst[i] = f32_to_bf16(src[i]);
}

// dst[C,R] = src[R,C]^T (bf16)
__global__ void cvt_transpose_bf16(const float* __restrict__ src,
                                   unsigned short* __restrict__ dst, int R, int C) {
    int i = blockIdx.x * 256 + threadIdx.x;
    if (i >= R * C) return;
    int r = i / C, c = i - r * C;
    dst[(size_t)c * R + r] = f32_to_bf16(src[i]);
}

__global__ void zero_u32(unsigned int* p, int n) {
    int i = blockIdx.x * 256 + threadIdx.x;
    if (i < n) p[i] = 0u;
}

// ---------------- Bahdanau attention + context + x assembly ----------------
// one block (256 threads) per batch row
__global__ void attention_kernel(const float* __restrict__ att1,   // [B*L, ATT]
                                 const float* __restrict__ att2,   // [B, ATT]
                                 const float* __restrict__ enc,    // [B*L, DEC]
                                 const float* __restrict__ W_fa,   // [ATT]
                                 const float* __restrict__ b_fa,   // [1]
                                 const float* __restrict__ emb,    // [V, EMB]
                                 const int*   __restrict__ captions, // [B, T]
                                 unsigned short* __restrict__ x_bf, // [B, XIN] bf16
                                 int t)
{
    __shared__ float red[256];
    __shared__ float sc[64];
    int b   = blockIdx.x;
    int tid = threadIdx.x;

    float a2 = att2[b * ATT + tid];
    float wf = W_fa[tid];

    for (int l = 0; l < L; ++l) {
        float v = tanhf(att1[((size_t)(b * L + l)) * ATT + tid] + a2) * wf;
        red[tid] = v;
        __syncthreads();
        for (int s = 128; s > 0; s >>= 1) {
            if (tid < s) red[tid] += red[tid + s];
            __syncthreads();
        }
        if (tid == 0) sc[l] = red[0] + b_fa[0];
        __syncthreads();
    }

    if (tid == 0) {
        float m = sc[0];
        for (int l = 1; l < L; ++l) m = fmaxf(m, sc[l]);
        float s = 0.0f;
        for (int l = 0; l < L; ++l) { sc[l] = expf(sc[l] - m); s += sc[l]; }
        float inv = 1.0f / s;
        for (int l = 0; l < L; ++l) sc[l] *= inv;
    }
    __syncthreads();

    int cap = captions[b * T + t];
    for (int d = tid; d < DEC; d += 256) {
        float acc = 0.0f;
        for (int l = 0; l < L; ++l)
            acc += sc[l] * enc[((size_t)(b * L + l)) * DEC + d];
        x_bf[(size_t)b * XIN + EMB + d] = f32_to_bf16(acc);                        // context
        x_bf[(size_t)b * XIN + d]       = f32_to_bf16(emb[(size_t)cap * EMB + d]); // embedding
    }
}

// ---------------- GRU gate fusion ----------------
__global__ void gru_gate_kernel(const float* __restrict__ gi,  // [B, 3*DEC]
                                const float* __restrict__ gh,  // [B, 3*DEC]
                                float* __restrict__ h,         // [B, DEC]
                                unsigned short* __restrict__ h_bf)
{
    int idx = blockIdx.x * 256 + threadIdx.x;
    if (idx >= B * DEC) return;
    int b = idx >> 9;          // /512
    int d = idx & 511;
    const float* gib = gi + (size_t)b * G3;
    const float* ghb = gh + (size_t)b * G3;
    float r = 1.0f / (1.0f + expf(-(gib[d]            + ghb[d])));
    float z = 1.0f / (1.0f + expf(-(gib[DEC + d]      + ghb[DEC + d])));
    float n = tanhf(gib[2 * DEC + d] + r * ghb[2 * DEC + d]);
    float hn = (1.0f - z) * n + z * h[idx];
    h[idx]    = hn;
    h_bf[idx] = f32_to_bf16(hn);
}

// ---------------- host orchestration ----------------
static inline void launch_gemm(const unsigned short* A, int lda,
                               const unsigned short* Bt, int ldb,
                               const float* bias, float* C, long long ldc,
                               int M, int N, int K, hipStream_t stream) {
    int tiles_m = M / 32;                 // 32-row macro-tiles
    int tiles_n = N / 16;
    int tiles = tiles_m * tiles_n;
    int blocks = (tiles + 7) / 8;
    wmma_gemm_bf16<<<blocks, 256, 0, stream>>>(A, lda, Bt, ldb, bias, C, ldc,
                                               tiles_m, tiles_n, K);
}

extern "C" void kernel_launch(void* const* d_in, const int* in_sizes, int n_in,
                              void* d_out, int out_size, void* d_ws, size_t ws_size,
                              hipStream_t stream) {
    const float* spatial = (const float*)d_in[0];
    const int*   caps    = (const int*)  d_in[1];
    const float* W_feat  = (const float*)d_in[2];
    const float* b_feat  = (const float*)d_in[3];
    const float* W_ea    = (const float*)d_in[4];
    const float* b_ea    = (const float*)d_in[5];
    const float* W_da    = (const float*)d_in[6];
    const float* b_da    = (const float*)d_in[7];
    const float* W_fa    = (const float*)d_in[8];
    const float* b_fa    = (const float*)d_in[9];
    const float* emb     = (const float*)d_in[10];
    const float* W_ih    = (const float*)d_in[11];
    const float* W_hh    = (const float*)d_in[12];
    const float* b_ih    = (const float*)d_in[13];
    const float* b_hh    = (const float*)d_in[14];
    const float* W_fc    = (const float*)d_in[15];
    const float* b_fc    = (const float*)d_in[16];
    float* out = (float*)d_out;

    // workspace carve-out (256B aligned)
    char* ws = (char*)d_ws;
    size_t off = 0;
    auto alloc = [&](size_t bytes) -> char* {
        char* p = ws + off;
        off += (bytes + 255) & ~(size_t)255;
        return p;
    };
    unsigned short* spatial_bf = (unsigned short*)alloc((size_t)BL * ENC * 2);
    unsigned short* Wt_feat    = (unsigned short*)alloc((size_t)DEC * ENC * 2);
    unsigned short* Wt_ea      = (unsigned short*)alloc((size_t)ATT * DEC * 2);
    unsigned short* Wt_da      = (unsigned short*)alloc((size_t)ATT * DEC * 2);
    unsigned short* Wih_bf     = (unsigned short*)alloc((size_t)G3 * XIN * 2);
    unsigned short* Whh_bf     = (unsigned short*)alloc((size_t)G3 * DEC * 2);
    unsigned short* Wt_fc      = (unsigned short*)alloc((size_t)V * DEC * 2);
    float*          enc        = (float*)alloc((size_t)BL * DEC * 4);
    unsigned short* enc_bf     = (unsigned short*)alloc((size_t)BL * DEC * 2);
    float*          att1       = (float*)alloc((size_t)BL * ATT * 4);
    float*          att2       = (float*)alloc((size_t)B * ATT * 4);
    float*          h          = (float*)alloc((size_t)B * DEC * 4);
    unsigned short* h_bf       = (unsigned short*)alloc((size_t)B * DEC * 2);
    unsigned short* x_bf       = (unsigned short*)alloc((size_t)B * XIN * 2);
    float*          gi         = (float*)alloc((size_t)B * G3 * 4);
    float*          gh         = (float*)alloc((size_t)B * G3 * 4);
    (void)ws_size; (void)in_sizes; (void)n_in; (void)out_size;

    auto grid1 = [](int n) { return dim3((unsigned)((n + 255) / 256)); };

    // --- one-time conversions (re-done every call for determinism) ---
    cvt_copy_bf16<<<grid1(BL * ENC), 256, 0, stream>>>(spatial, spatial_bf, BL * ENC);
    cvt_copy_bf16<<<grid1(G3 * XIN), 256, 0, stream>>>(W_ih, Wih_bf, G3 * XIN);
    cvt_copy_bf16<<<grid1(G3 * DEC), 256, 0, stream>>>(W_hh, Whh_bf, G3 * DEC);
    cvt_transpose_bf16<<<grid1(ENC * DEC), 256, 0, stream>>>(W_feat, Wt_feat, ENC, DEC);
    cvt_transpose_bf16<<<grid1(DEC * ATT), 256, 0, stream>>>(W_ea, Wt_ea, DEC, ATT);
    cvt_transpose_bf16<<<grid1(DEC * ATT), 256, 0, stream>>>(W_da, Wt_da, DEC, ATT);
    cvt_transpose_bf16<<<grid1(DEC * V), 256, 0, stream>>>(W_fc, Wt_fc, DEC, V);
    zero_u32<<<grid1(B * DEC), 256, 0, stream>>>((unsigned int*)h, B * DEC);
    zero_u32<<<grid1(B * DEC / 2), 256, 0, stream>>>((unsigned int*)h_bf, B * DEC / 2);

    // --- loop-invariant encoder projections ---
    // enc = spatial @ W_feat + b_feat   [3136,2048]x[2048,512]
    launch_gemm(spatial_bf, ENC, Wt_feat, ENC, b_feat, enc, DEC, BL, DEC, ENC, stream);
    cvt_copy_bf16<<<grid1(BL * DEC), 256, 0, stream>>>(enc, enc_bf, BL * DEC);
    // att1 = enc @ W_ea + b_ea          [3136,512]x[512,256]
    launch_gemm(enc_bf, DEC, Wt_ea, DEC, b_ea, att1, ATT, BL, ATT, DEC, stream);

    // --- recurrent decode: 50 steps, 6 kernels each (graph-captured) ---
    for (int t = 0; t < T; ++t) {
        // att2 = h @ W_da + b_da        [64,512]x[512,256]
        launch_gemm(h_bf, DEC, Wt_da, DEC, b_da, att2, ATT, B, ATT, DEC, stream);
        // scores/softmax/context + x = [emb_t ; context]
        attention_kernel<<<B, 256, 0, stream>>>(att1, att2, enc, W_fa, b_fa,
                                                emb, caps, x_bf, t);
        // gi = x @ W_ih^T + b_ih        [64,1024]x[1024,1536]
        launch_gemm(x_bf, XIN, Wih_bf, XIN, b_ih, gi, G3, B, G3, XIN, stream);
        // gh = h @ W_hh^T + b_hh        [64,512]x[512,1536]
        launch_gemm(h_bf, DEC, Whh_bf, DEC, b_hh, gh, G3, B, G3, DEC, stream);
        // gate fusion -> h, h_bf
        gru_gate_kernel<<<grid1(B * DEC), 256, 0, stream>>>(gi, gh, h, h_bf);
        // logits_t = h @ W_fc + b_fc    [64,512]x[512,10000] -> out[:, t, :]
        launch_gemm(h_bf, DEC, Wt_fc, DEC, b_fc,
                    out + (size_t)t * V, (long long)T * V, B, V, DEC, stream);
    }
}
